// TitanBlock_76244259438858
// MI455X (gfx1250) — compile-verified, matrix operand order checked by reading, not run on hardware
//
#include <hip/hip_runtime.h>
#include <hip/hip_bf16.h>
#include <stddef.h>
#include <stdint.h>

// ---------------------------------------------------------------------------
// TitanBlock for MI455X (gfx1250, wave32, WMMA 16x16x32 bf16).
// ~224 GFLOP, working set << 192MB L2 -> matrix-core bound.
// GEMM v2: W tile double-buffered in LDS via async global->LDS copies
// (ASYNCcnt / s_wait_asynccnt).
// ---------------------------------------------------------------------------

#define B_  4
#define L_  2048
#define D_  1024
#define BL_ (B_ * L_)
#define BLD ((size_t)B_ * L_ * D_)
#define BLL ((size_t)B_ * L_ * L_)
#define DD  ((size_t)D_ * D_)

typedef __bf16 bf16;
typedef bf16  v8bf  __attribute__((ext_vector_type(8)));
typedef bf16  v16bf __attribute__((ext_vector_type(16)));
typedef float v8f   __attribute__((ext_vector_type(8)));
typedef int   i32x4 __attribute__((ext_vector_type(4)));

#define ACT_NONE    0
#define ACT_SILU    1
#define ACT_SIGMOID 2

#if __has_builtin(__builtin_amdgcn_global_load_async_to_lds_b128) && \
    __has_builtin(__builtin_amdgcn_s_wait_asynccnt)
#define HAVE_ASYNC_LDS 1
#else
#define HAVE_ASYNC_LDS 0
#endif

// The async builtin wants i32x4* in explicit address spaces (1=global, 3=LDS).
// Generic LDS pointer low 32 bits are the LDS offset; generic global == as(1).
#define GADDR(p) ((__attribute__((address_space(1))) i32x4*)(uintptr_t)(p))
#define LADDR(p) ((__attribute__((address_space(3))) i32x4*)(unsigned)(uintptr_t)(p))

// LDS W-tile row pitch in bf16 elements: 80B = 20 banks -> the 16 lanes'
// ds_load_b128 starts land on 16 distinct 4-bank groups (no conflicts).
#define LDSROW 40

// ---------------------------------------------------------------------------
// WMMA fragment loaders. 16-bit A (16x32) layout per CDNA5 ISA 7.12.2:
//   lanes 0-15 : row = lane,    elems 0-7 = K 0..7,  8-15 = K 16..23
//   lanes 16-31: row = lane-16, elems 0-7 = K 8..15, 8-15 = K 24..31
// B (32x16) is symmetric with N in the lane slot.
// ---------------------------------------------------------------------------
__device__ __forceinline__ v16bf load_frag(const bf16* __restrict__ base,
                                           int stride, int k0, int lane) {
    const int r  = lane & 15;
    const int kb = k0 + ((lane >> 4) << 3);
    const bf16* p = base + (size_t)r * stride + kb;
    v8bf lo = *(const v8bf*)p;
    v8bf hi = *(const v8bf*)(p + 16);
    return __builtin_shufflevector(lo, hi, 0, 1, 2, 3, 4, 5, 6, 7,
                                           8, 9, 10, 11, 12, 13, 14, 15);
}

__device__ __forceinline__ v16bf load_frag_lds(const bf16* base, int lane) {
    const int r  = lane & 15;
    const int kb = (lane >> 4) << 3;
    const bf16* p = base + r * LDSROW + kb;
    v8bf lo = *(const v8bf*)p;
    v8bf hi = *(const v8bf*)(p + 16);
    return __builtin_shufflevector(lo, hi, 0, 1, 2, 3, 4, 5, 6, 7,
                                           8, 9, 10, 11, 12, 13, 14, 15);
}

// stage one 64(N) x 32(K) bf16 W tile into LDS: 256 threads x one b128 each
__device__ __forceinline__ void stage_tile(bf16* ldsbuf,
                                           const bf16* __restrict__ Wg,
                                           int Kd, int k0, int tid) {
    const int row = tid >> 2;      // 0..63
    const int seg = tid & 3;       // 4 x 8 bf16 = 32 K elems
    const bf16* src = Wg + (size_t)row * Kd + k0 + seg * 8;
    bf16* dst = ldsbuf + row * LDSROW + seg * 8;
#if HAVE_ASYNC_LDS
    __builtin_amdgcn_global_load_async_to_lds_b128(GADDR(src), LADDR(dst), 0, 0);
#else
    *(v8bf*)dst = *(const v8bf*)src;
#endif
}

// C = act((A @ W^T) * scale + bias) [* mulF] [+ addF]
// Block = 256 threads = 8 waves; block tile 128(M) x 64(N); wave tile 16x64.
// M, N multiples of 128/64 and Kd multiple of 32 (true for every call here).
__global__ __launch_bounds__(256)
void k_gemm(const bf16* __restrict__ A, const bf16* __restrict__ W,
            const float* __restrict__ bias,
            float* __restrict__ outF, bf16* __restrict__ outB,
            const float* __restrict__ mulF, const float* __restrict__ addF,
            int M, int N, int Kd, float scale, int act) {
    __shared__ bf16 wtile[2][64 * LDSROW];
    const int tid   = threadIdx.x;
    const int lane  = tid & 31;
    const int wave  = tid >> 5;
    const int mTile = blockIdx.y * 128 + wave * 16;
    const int nBase = blockIdx.x * 64;

    const bf16* Ab = A + (size_t)mTile * Kd;
    const bf16* Wg = W + (size_t)nBase * Kd;
    v8f acc[4] = {};

    stage_tile(wtile[0], Wg, Kd, 0, tid);
    int buf = 0;
    for (int k0 = 0; k0 < Kd; k0 += 32) {
        const bool more = (k0 + 32) < Kd;
        if (more) {
            stage_tile(wtile[buf ^ 1], Wg, Kd, k0 + 32, tid);
#if HAVE_ASYNC_LDS
            __builtin_amdgcn_s_wait_asynccnt(1);   // buf's copy done, next in flight
#endif
        } else {
#if HAVE_ASYNC_LDS
            __builtin_amdgcn_s_wait_asynccnt(0);
#endif
        }
        __syncthreads();                           // tile visible to all 8 waves

        // speculative prefetch of the next A slab (global_prefetch_b8)
        __builtin_prefetch(Ab + (size_t)(lane & 15) * Kd + k0 + 32, 0, 1);
        v16bf a = load_frag(Ab, Kd, k0, lane);
#pragma unroll
        for (int j = 0; j < 4; ++j) {
            v16bf b = load_frag_lds(&wtile[buf][(j * 16) * LDSROW], lane);
            acc[j] = __builtin_amdgcn_wmma_f32_16x16x32_bf16(
                false, a, false, b, (short)0, acc[j], false, false);
        }
        __syncthreads();                           // reads done before restage
        buf ^= 1;
    }

    // C/D layout: VGPR r -> M = r (lanes 0-15) / r+8 (lanes 16-31); N = lane&15
    const int nc = lane & 15;
    const int mh = (lane >> 4) << 3;
#pragma unroll
    for (int j = 0; j < 4; ++j) {
        const int n = nBase + j * 16 + nc;
        const float bv = bias ? bias[n] : 0.0f;
#pragma unroll
        for (int r = 0; r < 8; ++r) {
            const int m = mTile + r + mh;
            const size_t idx = (size_t)m * N + n;
            float v = acc[j][r] * scale + bv;
            if (act == ACT_SILU)         v = v / (1.0f + __expf(-v));
            else if (act == ACT_SIGMOID) v = 1.0f / (1.0f + __expf(-v));
            if (mulF) v *= mulF[idx];
            if (addF) v += addF[idx];
            if (outF) outF[idx] = v;
            if (outB) outB[idx] = (bf16)v;
        }
    }
}

// ---------------------------------------------------------------------------
// Elementwise / reduction helpers
// ---------------------------------------------------------------------------
__global__ void k_f32_to_bf16(const float* __restrict__ s, bf16* __restrict__ d,
                              size_t n) {
    size_t i = (size_t)blockIdx.x * blockDim.x + threadIdx.x;
    if (i < n) d[i] = (bf16)s[i];
}

// dst[e*D + d] = src[d*D + e]  (mem_w pre-transpose, f32 -> bf16)
__global__ void k_transpose_cvt(const float* __restrict__ s,
                                bf16* __restrict__ d) {
    size_t i = (size_t)blockIdx.x * blockDim.x + threadIdx.x;
    if (i < DD) {
        int e = (int)(i / D_), c = (int)(i % D_);
        d[i] = (bf16)s[(size_t)c * D_ + e];
    }
}

__global__ void k_zero1(float* p) { p[0] = 0.0f; }

// depthwise conv over L, kernel 3, zero pad 1, + bias; f32 in, bf16 out
__global__ void k_dwconv(const float* __restrict__ x,
                         const float* __restrict__ dw,
                         const float* __restrict__ db,
                         bf16* __restrict__ y) {
    size_t idx = (size_t)blockIdx.x * blockDim.x + threadIdx.x;
    if (idx >= BLD) return;
    int c = (int)(idx & (D_ - 1));
    size_t l = (idx >> 10) & (L_ - 1);
    float xm = (l > 0)      ? x[idx - D_] : 0.0f;
    float x0 = x[idx];
    float xp = (l < L_ - 1) ? x[idx + D_] : 0.0f;
    y[idx] = (bf16)(db[c] + dw[c * 3 + 0] * xm + dw[c * 3 + 1] * x0 +
                    dw[c * 3 + 2] * xp);
}

__device__ __forceinline__ float block_sum(float v, float* red) {
    int t = threadIdx.x;
    red[t] = v; __syncthreads();
    for (int s = 128; s > 0; s >>= 1) {
        if (t < s) red[t] += red[t + s];
        __syncthreads();
    }
    float r = red[0]; __syncthreads();
    return r;
}
__device__ __forceinline__ float block_max(float v, float* red) {
    int t = threadIdx.x;
    red[t] = v; __syncthreads();
    for (int s = 128; s > 0; s >>= 1) {
        if (t < s) red[t] = fmaxf(red[t], red[t + s]);
        __syncthreads();
    }
    float r = red[0]; __syncthreads();
    return r;
}

// one block per row of D_ elements: y = x / max(||x||, 1e-12) -> bf16
__global__ __launch_bounds__(256)
void k_l2norm(const float* __restrict__ X, bf16* __restrict__ Y) {
    __shared__ float red[256];
    const float* x = X + (size_t)blockIdx.x * D_;
    bf16* y = Y + (size_t)blockIdx.x * D_;
    float s = 0.0f;
    for (int i = threadIdx.x; i < D_; i += 256) { float v = x[i]; s += v * v; }
    s = block_sum(s, red);
    float inv = 1.0f / fmaxf(sqrtf(s), 1e-12f);
    for (int i = threadIdx.x; i < D_; i += 256) y[i] = (bf16)(x[i] * inv);
}

// one block per row of N cols: P = softmax(S) -> bf16 (scale pre-applied)
__global__ __launch_bounds__(256)
void k_softmax(const float* __restrict__ S, bf16* __restrict__ P, int N) {
    __shared__ float red[256];
    const float* s = S + (size_t)blockIdx.x * N;
    bf16* p = P + (size_t)blockIdx.x * N;
    float m = -1e30f;
    for (int i = threadIdx.x; i < N; i += 256) m = fmaxf(m, s[i]);
    m = block_max(m, red);
    float sum = 0.0f;
    for (int i = threadIdx.x; i < N; i += 256) sum += __expf(s[i] - m);
    sum = block_sum(sum, red);
    float inv = 1.0f / sum;
    for (int i = threadIdx.x; i < N; i += 256)
        p[i] = (bf16)(__expf(s[i] - m) * inv);
}

// bf16 [R x C] -> [C x R] per batch (blockIdx.z), 32x32 LDS tiles
__global__ __launch_bounds__(256)
void k_transpose_bf16(const bf16* __restrict__ src, bf16* __restrict__ dst,
                      int R, int C) {
    __shared__ bf16 tile[32][33];
    size_t boff = (size_t)blockIdx.z * R * C;
    int r0 = blockIdx.y * 32, c0 = blockIdx.x * 32;
    int tx = threadIdx.x & 31, ty = threadIdx.x >> 5;
    for (int i = 0; i < 32; i += 8)
        tile[ty + i][tx] = src[boff + (size_t)(r0 + ty + i) * C + c0 + tx];
    __syncthreads();
    for (int i = 0; i < 32; i += 8)
        dst[boff + (size_t)(c0 + ty + i) * R + r0 + tx] = tile[tx][ty + i];
}

// one block per row: layernorm over D_, eps 1e-5
__global__ __launch_bounds__(256)
void k_layernorm(const float* __restrict__ X, const float* __restrict__ g,
                 const float* __restrict__ b, float* __restrict__ out) {
    __shared__ float red[256];
    const float* x = X + (size_t)blockIdx.x * D_;
    float* o = out + (size_t)blockIdx.x * D_;
    float s = 0.0f, s2 = 0.0f;
    for (int i = threadIdx.x; i < D_; i += 256) {
        float v = x[i]; s += v; s2 += v * v;
    }
    s  = block_sum(s, red);
    s2 = block_sum(s2, red);
    float mu  = s / D_;
    float var = s2 / D_ - mu * mu;
    float rs  = rsqrtf(var + 1e-5f);
    for (int i = threadIdx.x; i < D_; i += 256)
        o[i] = (x[i] - mu) * rs * g[i] + b[i];
}

__global__ __launch_bounds__(256)
void k_mse(const float* __restrict__ p, const float* __restrict__ t,
           float* __restrict__ out, size_t n) {
    __shared__ float red[256];
    float s = 0.0f;
    for (size_t i = (size_t)blockIdx.x * 256 + threadIdx.x; i < n;
         i += (size_t)gridDim.x * 256) {
        float d = p[i] - t[i]; s += d * d;
    }
    s = block_sum(s, red);
    if (threadIdx.x == 0) atomicAdd(out, s / (float)n);
}

// ---------------------------------------------------------------------------
// Launcher
// ---------------------------------------------------------------------------
extern "C" void kernel_launch(void* const* d_in, const int* in_sizes, int n_in,
                              void* d_out, int out_size, void* d_ws,
                              size_t ws_size, hipStream_t stream) {
    const float* x     = (const float*)d_in[0];
    const float* wq    = (const float*)d_in[1];  const float* bq    = (const float*)d_in[2];
    const float* wk    = (const float*)d_in[3];  const float* bk    = (const float*)d_in[4];
    const float* wv    = (const float*)d_in[5];  const float* bv    = (const float*)d_in[6];
    const float* dwq   = (const float*)d_in[7];  const float* dwq_b = (const float*)d_in[8];
    const float* pwq   = (const float*)d_in[9];  const float* pwq_b = (const float*)d_in[10];
    const float* dwk   = (const float*)d_in[11]; const float* dwk_b = (const float*)d_in[12];
    const float* pwk   = (const float*)d_in[13]; const float* pwk_b = (const float*)d_in[14];
    const float* dwv   = (const float*)d_in[15]; const float* dwv_b = (const float*)d_in[16];
    const float* pwv   = (const float*)d_in[17]; const float* pwv_b = (const float*)d_in[18];
    const float* wg    = (const float*)d_in[19]; const float* bg    = (const float*)d_in[20];
    const float* wo    = (const float*)d_in[21]; const float* bo    = (const float*)d_in[22];
    const float* mem_w = (const float*)d_in[23];
    const float* ln_g  = (const float*)d_in[24]; const float* ln_b  = (const float*)d_in[25];
    float* out = (float*)d_out;

    size_t off = 0;
    auto alloc = [&](size_t bytes) -> void* {
        void* p = (char*)d_ws + off;
        off += (bytes + 255) & ~(size_t)255;
        return p;
    };
    bf16* xb    = (bf16*)alloc(BLD * 2);
    bf16* wqb   = (bf16*)alloc(DD * 2);
    bf16* wkb   = (bf16*)alloc(DD * 2);
    bf16* wvb   = (bf16*)alloc(DD * 2);
    bf16* pwqb  = (bf16*)alloc(DD * 2);
    bf16* pwkb  = (bf16*)alloc(DD * 2);
    bf16* pwvb  = (bf16*)alloc(DD * 2);
    bf16* wgb   = (bf16*)alloc(DD * 2);
    bf16* wob   = (bf16*)alloc(DD * 2);
    bf16* memwT = (bf16*)alloc(DD * 2);
    float* Qf   = (float*)alloc(BLD * 4);
    float* Kf   = (float*)alloc(BLD * 4);
    float* Vf   = (float*)alloc(BLD * 4);
    bf16* Qc    = (bf16*)alloc(BLD * 2);
    bf16* Kc    = (bf16*)alloc(BLD * 2);
    bf16* Vc    = (bf16*)alloc(BLD * 2);
    bf16* Vb    = (bf16*)alloc(BLD * 2);
    bf16* Qnb   = (bf16*)alloc(BLD * 2);
    bf16* Knb   = (bf16*)alloc(BLD * 2);
    float* S    = (float*)alloc(BLL * 4);
    bf16* P     = (bf16*)alloc(BLL * 2);
    bf16* Vt    = (bf16*)alloc(BLD * 2);
    float* attF = (float*)alloc(BLD * 4);
    bf16* attB  = (bf16*)alloc(BLD * 2);
    bf16* gated = (bf16*)alloc(BLD * 2);
    float* oPre = (float*)alloc(BLD * 4);
    float* mPrd = S;  // scores dead after softmax; reuse for mem_pred

    auto gemm = [&](const bf16* A, const bf16* Wm, const float* bias,
                    float* oF, bf16* oB, const float* mF, const float* aF,
                    int M, int N, int Kd, float scale, int act) {
        dim3 g(N / 64, (M + 127) / 128);
        k_gemm<<<g, 256, 0, stream>>>(A, Wm, bias, oF, oB, mF, aF,
                                      M, N, Kd, scale, act);
    };

    const int EB = (int)((BLD + 255) / 256);

    // 0) init scalar output; convert weights & input to bf16
    k_zero1<<<1, 1, 0, stream>>>(out + BLD);
    k_f32_to_bf16<<<(int)((DD + 255) / 256), 256, 0, stream>>>(wq,  wqb,  DD);
    k_f32_to_bf16<<<(int)((DD + 255) / 256), 256, 0, stream>>>(wk,  wkb,  DD);
    k_f32_to_bf16<<<(int)((DD + 255) / 256), 256, 0, stream>>>(wv,  wvb,  DD);
    k_f32_to_bf16<<<(int)((DD + 255) / 256), 256, 0, stream>>>(pwq, pwqb, DD);
    k_f32_to_bf16<<<(int)((DD + 255) / 256), 256, 0, stream>>>(pwk, pwkb, DD);
    k_f32_to_bf16<<<(int)((DD + 255) / 256), 256, 0, stream>>>(pwv, pwvb, DD);
    k_f32_to_bf16<<<(int)((DD + 255) / 256), 256, 0, stream>>>(wg,  wgb,  DD);
    k_f32_to_bf16<<<(int)((DD + 255) / 256), 256, 0, stream>>>(wo,  wob,  DD);
    k_transpose_cvt<<<(int)((DD + 255) / 256), 256, 0, stream>>>(mem_w, memwT);
    k_f32_to_bf16<<<EB, 256, 0, stream>>>(x, xb, BLD);

    // 1) QKV projections with fused SiLU
    gemm(xb, wqb, bq, Qf, nullptr, nullptr, nullptr, BL_, D_, D_, 1.f, ACT_SILU);
    gemm(xb, wkb, bk, Kf, nullptr, nullptr, nullptr, BL_, D_, D_, 1.f, ACT_SILU);
    gemm(xb, wvb, bv, Vf, nullptr, nullptr, nullptr, BL_, D_, D_, 1.f, ACT_SILU);

    // 2) depthwise conv (K=3) + bias -> bf16
    k_dwconv<<<EB, 256, 0, stream>>>(Qf, dwq, dwq_b, Qc);
    k_dwconv<<<EB, 256, 0, stream>>>(Kf, dwk, dwk_b, Kc);
    k_dwconv<<<EB, 256, 0, stream>>>(Vf, dwv, dwv_b, Vc);

    // 3) pointwise 1x1 conv = GEMM (+bias); V also needs bf16 copy for attn
    gemm(Qc, pwqb, pwq_b, Qf, nullptr, nullptr, nullptr, BL_, D_, D_, 1.f, ACT_NONE);
    gemm(Kc, pwkb, pwk_b, Kf, nullptr, nullptr, nullptr, BL_, D_, D_, 1.f, ACT_NONE);
    gemm(Vc, pwvb, pwv_b, Vf, Vb,      nullptr, nullptr, BL_, D_, D_, 1.f, ACT_NONE);

    // 4) L2-normalize Q, K rows
    k_l2norm<<<BL_, 256, 0, stream>>>(Qf, Qnb);
    k_l2norm<<<BL_, 256, 0, stream>>>(Kf, Knb);

    // 5) attention: S = Qn Kn^T / 32 ; P = softmax(S) ; attn = P V
    for (int b = 0; b < B_; ++b)
        gemm(Qnb + (size_t)b * L_ * D_, Knb + (size_t)b * L_ * D_, nullptr,
             S + (size_t)b * L_ * L_, nullptr, nullptr, nullptr,
             L_, L_, D_, 0.03125f, ACT_NONE);
    k_softmax<<<BL_, 256, 0, stream>>>(S, P, L_);
    k_transpose_bf16<<<dim3(D_ / 32, L_ / 32, B_), 256, 0, stream>>>(Vb, Vt, L_, D_);
    for (int b = 0; b < B_; ++b)
        gemm(P + (size_t)b * L_ * L_, Vt + (size_t)b * D_ * L_, nullptr,
             attF + (size_t)b * L_ * D_, attB + (size_t)b * L_ * D_,
             nullptr, nullptr, L_, D_, L_, 1.f, ACT_NONE);

    // 6) gate: gated = sigmoid(attn Wg^T + bg) * attn
    gemm(attB, wgb, bg, nullptr, gated, attF, nullptr, BL_, D_, D_, 1.f, ACT_SIGMOID);

    // 7) output proj + residual, then layernorm into d_out
    gemm(gated, wob, bo, oPre, nullptr, nullptr, x, BL_, D_, D_, 1.f, ACT_NONE);
    k_layernorm<<<BL_, 256, 0, stream>>>(oPre, ln_g, ln_b, out);

    // 8) mem branch: mem_pred = Kn @ mem_w ; mem_loss = mean((pred - V)^2)
    gemm(Knb, memwT, nullptr, mPrd, nullptr, nullptr, nullptr, BL_, D_, D_, 1.f, ACT_NONE);
    k_mse<<<1024, 256, 0, stream>>>(mPrd, Vf, out + BLD, BLD);
}
